// MeanField_45707041964164
// MI455X (gfx1250) — matrix operand order repaired and checked by reading, not compiled, not run on hardware
//
#include <hip/hip_runtime.h>
#include <hip/hip_bf16.h>

#define NN 8192
#define ITERS 50
#define EPSF 1.1920928955078125e-07f
#define LOG_HALF -0.6931471805599453f

typedef __attribute__((ext_vector_type(16))) __bf16 v16bf;
typedef __attribute__((ext_vector_type(8)))  __bf16 v8bf;
typedef __attribute__((ext_vector_type(8)))  float  v8f;

// ---------------------------------------------------------------------------
// Init: log_prob = log(0.5), tm1 = exp(log 0.5) - 1 = -0.5, diag = J[i,i]+eps
// ---------------------------------------------------------------------------
__global__ void mf_init(const float* __restrict__ J, float* lp0, float* lp1,
                        __bf16* tm1b, float* tm1f, float* diag) {
  int i = blockIdx.x * blockDim.x + threadIdx.x;
  if (i < NN) {
    lp0[i] = LOG_HALF;
    lp1[i] = LOG_HALF;
    tm1b[i] = (__bf16)(-0.5f);
    tm1f[i] = -0.5f;
    diag[i] = J[(size_t)i * NN + i] + EPSF;
  }
}

// ---------------------------------------------------------------------------
// One-time J fp32 -> bf16 conversion (268MB -> 134MB; fits in 192MB L2)
// ---------------------------------------------------------------------------
__global__ void mf_convert(const float* __restrict__ J, __bf16* __restrict__ Jb) {
  size_t i = ((size_t)blockIdx.x * blockDim.x + threadIdx.x) * 8;
  const float4* p = (const float4*)(J + i);
  float4 f0 = p[0];
  float4 f1 = p[1];
  v8bf o;
  o[0] = (__bf16)f0.x; o[1] = (__bf16)f0.y; o[2] = (__bf16)f0.z; o[3] = (__bf16)f0.w;
  o[4] = (__bf16)f1.x; o[5] = (__bf16)f1.y; o[6] = (__bf16)f1.z; o[7] = (__bf16)f1.w;
  *(v8bf*)(Jb + i) = o;
}

// ---------------------------------------------------------------------------
// Matvec y = J_bf16 @ tm1 via v_wmma_f32_16x16x32_bf16.
// Block = 128 threads = 4 waves; block rb owns rows [rb*16, rb*16+16);
// each wave covers a K-segment of 2048 (64 WMMAs); LDS reduce across waves.
// B trick: ALL lanes load the same tm1 slice, so B = tm1 broadcast into all
// 16 columns. Every column of D then equals y — we read column 0. This keeps
// the loop fully convergent: no exec masking, no zero-fill of unused columns,
// and B loads are lane-uniform broadcasts that stay hot in L0.
// ---------------------------------------------------------------------------
__global__ void __launch_bounds__(128)
mf_matvec_bf16(const __bf16* __restrict__ Jb, const __bf16* __restrict__ tm1b,
               float* __restrict__ y) {
  __shared__ float red[4][16];
  const int rb   = blockIdx.x;            // 512 row blocks
  const int wid  = threadIdx.x >> 5;      // wave id 0..3
  const int lane = threadIdx.x & 31;
  const int m    = lane & 15;             // row within tile (A layout)
  const int half = lane >> 4;             // K-half select (A/B layout)
  const size_t rowbase = ((size_t)rb * 16 + m) * NN;
  const int kstart = wid * (NN / 4);

  // A: lane holds row m, K = half*8..+7 (lo) and 16+half*8..+7 (hi)
  const __bf16* ap = Jb + rowbase + kstart + half * 8;
  // B: lane holds K = half*8..+7 (lo) and 16+half*8..+7 (hi), matching A
  const __bf16* bp = tm1b + kstart + half * 8;

  v8f c = {};
#pragma unroll 4
  for (int t = 0; t < (NN / 4) / 32; ++t) {
    v8bf lo = *(const v8bf*)(ap);
    v8bf hi = *(const v8bf*)(ap + 16);
    v8bf b0 = *(const v8bf*)(bp);
    v8bf b1 = *(const v8bf*)(bp + 16);
    v16bf a, bm;
#pragma unroll
    for (int i = 0; i < 8; ++i) {
      a[i]      = lo[i];
      a[8 + i]  = hi[i];
      bm[i]     = b0[i];
      bm[8 + i] = b1[i];
    }
    c = __builtin_amdgcn_wmma_f32_16x16x32_bf16(false, a, false, bm,
                                                (short)0, c, false, false);
    ap += 32;
    bp += 32;
  }

  // D column 0: lane 0 -> rows 0..7 (c[0..7]), lane 16 -> rows 8..15
  if (m == 0) {
#pragma unroll
    for (int r = 0; r < 8; ++r) red[wid][half * 8 + r] = c[r];
  }
  __syncthreads();
  if (threadIdx.x < 16) {
    y[rb * 16 + threadIdx.x] = red[0][threadIdx.x] + red[1][threadIdx.x] +
                               red[2][threadIdx.x] + red[3][threadIdx.x];
  }
}

// ---------------------------------------------------------------------------
// Fallback fp32 matvec (one wave per row) if ws can't hold bf16 copy of J.
// ---------------------------------------------------------------------------
__global__ void __launch_bounds__(256)
mf_matvec_f32(const float* __restrict__ J, const float* __restrict__ tm1f,
              float* __restrict__ y) {
  const int row  = blockIdx.x * 8 + (threadIdx.x >> 5);
  const int lane = threadIdx.x & 31;
  const float* rp = J + (size_t)row * NN;
  float s = 0.0f;
  for (int k = lane * 4; k < NN; k += 32 * 4) {
    float4 v = *(const float4*)(rp + k);
    float4 t = *(const float4*)(tm1f + k);
    s += v.x * t.x + v.y * t.y + v.z * t.z + v.w * t.w;
  }
#pragma unroll
  for (int off = 16; off > 0; off >>= 1) s += __shfl_down(s, off, 32);
  if (lane == 0) y[row] = s;
}

// ---------------------------------------------------------------------------
// Pointwise update: tmp = log(1-t+eps) + 8*(J@tm1) - 4*diagJ*tm1 + 2b - 1,
// normalize [tmp, lp1] via logsumexp, emit prob_step column, prep next tm1.
// ---------------------------------------------------------------------------
__global__ void mf_update(float* lp0, float* lp1, const float* __restrict__ y,
                          const float* __restrict__ diag,
                          const float* __restrict__ b,
                          __bf16* tm1b, float* tm1f, float* out, int iter) {
  int i = blockIdx.x * blockDim.x + threadIdx.x;
  if (i >= NN) return;
  float l0 = lp0[i];
  float l1 = lp1[i];
  float t   = expf(l0);
  float tm1 = t - 1.0f;
  float tmp = logf(1.0f - t + EPSF) + 8.0f * y[i] - 4.0f * diag[i] * tm1 +
              2.0f * b[i] - 1.0f;
  float a0 = tmp, a1 = l1;
  float mx  = fmaxf(a0, a1);
  float lse = mx + logf(expf(a0 - mx) + expf(a1 - mx));
  float n0 = a0 - lse;
  float n1 = a1 - lse;
  lp0[i] = n0;
  lp1[i] = n1;
  float p = expf(n0);
  out[2 * NN + (size_t)i * ITERS + iter] = p;   // prob_step[i, iter]
  float nt = p - 1.0f;                          // tm1 for next iteration
  tm1b[i] = (__bf16)nt;
  tm1f[i] = nt;
  if (iter == ITERS - 1) {                      // exp(log_prob) [N,2]
    out[2 * i]     = p;
    out[2 * i + 1] = expf(n1);
  }
}

// ---------------------------------------------------------------------------
extern "C" void kernel_launch(void* const* d_in, const int* in_sizes, int n_in,
                              void* d_out, int out_size, void* d_ws, size_t ws_size,
                              hipStream_t stream) {
  const float* J = (const float*)d_in[0];
  const float* b = (const float*)d_in[1];
  // d_in[2] msg_node (int64), d_in[3] msg_adj, d_in[4] max_iter: unused.
  float* out = (float*)d_out;
  char* ws = (char*)d_ws;

  // Workspace layout (small buffers first so fallback works with tiny ws)
  size_t off = 0;
  auto take = [&](size_t bytes) {
    char* p = ws + off;
    off = (off + bytes + 255) & ~(size_t)255;
    return (void*)p;
  };
  __bf16* tm1b = (__bf16*)take((size_t)NN * 2);
  float*  tm1f = (float*)take((size_t)NN * 4);
  float*  lp0  = (float*)take((size_t)NN * 4);
  float*  lp1  = (float*)take((size_t)NN * 4);
  float*  y    = (float*)take((size_t)NN * 4);
  float*  diag = (float*)take((size_t)NN * 4);
  const size_t Jbytes = (size_t)NN * NN * 2;
  __bf16* Jb = (__bf16*)take(Jbytes);
  const bool use_bf16 = (off <= ws_size);

  mf_init<<<(NN + 255) / 256, 256, 0, stream>>>(J, lp0, lp1, tm1b, tm1f, diag);

  if (use_bf16) {
    const size_t nconv = (size_t)NN * NN / 8;   // 8 elems per thread
    mf_convert<<<(unsigned)(nconv / 256), 256, 0, stream>>>(J, Jb);
  }

  for (int iter = 0; iter < ITERS; ++iter) {
    if (use_bf16) {
      mf_matvec_bf16<<<NN / 16, 128, 0, stream>>>(Jb, tm1b, y);
    } else {
      mf_matvec_f32<<<NN / 8, 256, 0, stream>>>(J, tm1f, y);
    }
    mf_update<<<(NN + 255) / 256, 256, 0, stream>>>(lp0, lp1, y, diag, b,
                                                    tm1b, tm1f, out, iter);
  }
}
